// RegionProposalNetwork_52613349376111
// MI455X (gfx1250) — matrix operand order repaired and checked by reading, not compiled, not run on hardware
//
#include <hip/hip_runtime.h>
#include <cstdint>
#include <cstddef>

typedef __attribute__((ext_vector_type(16))) _Float16 v16h;
typedef __attribute__((ext_vector_type(8)))  float    v8f;
typedef __attribute__((ext_vector_type(4)))  int      v4i;

#define NPOS   2500      // 50*50 positions per batch
#define BATCH  8
#define NANCH  22500     // NPOS * 9
#define POST   300
#define NEGV   (-1.0e9f)

// ---------------- workspace layout (bytes) ----------------
#define OFF_PADF   0ull
#define SZ_PADF    (8ull*52*52*512*2)            // padded NHWC f16 feat
#define OFF_WHN    (OFF_PADF + SZ_PADF)
#define SZ_WHN     (512ull*4608*2)               // W1 repacked [oc][k] f16
#define OFF_CONV1  (OFF_WHN + SZ_WHN)
#define SZ_CONV1   (8ull*2500*512*2)             // conv1 out f16 [b*p][oc]
#define OFF_BOXES  (OFF_CONV1 + SZ_CONV1)
#define SZ_BOXES   (8ull*22500*4*4)              // decoded boxes f32
#define OFF_SCORES (OFF_BOXES + SZ_BOXES)        // masked fg scores f32

// -------- CDNA5 async global->LDS path (guarded; sync fallback) --------
#if defined(__has_builtin)
#if __has_builtin(__builtin_amdgcn_global_load_async_to_lds_b128) && \
    __has_builtin(__builtin_amdgcn_s_wait_asynccnt)
#define USE_ASYNC_LDS 1
#endif
#endif
#ifndef USE_ASYNC_LDS
#define USE_ASYNC_LDS 0
#endif

#if USE_ASYNC_LDS
// builtin signature (from hipcc diagnostic): (int4 AS1*, int4 AS3*, imm, imm)
#define AS1P(p) ((__attribute__((address_space(1))) v4i*)(p))
#define AS3P(p) ((__attribute__((address_space(3))) v4i*)(p))
#endif

// =============== kernel 1: pad + convert feat to NHWC f16 ===============
__global__ void pad_nhwc_f16(const float* __restrict__ feat,
                             _Float16* __restrict__ padf) {
    size_t i = (size_t)blockIdx.x * blockDim.x + threadIdx.x;
    const size_t N = (size_t)BATCH * 52 * 52 * 512;
    if (i >= N) return;
    int c = (int)(i & 511);
    size_t r = i >> 9;
    int px = (int)(r % 52); r /= 52;
    int py = (int)(r % 52);
    int b  = (int)(r / 52);
    float v = 0.0f;
    if (py >= 1 && py <= 50 && px >= 1 && px <= 50)
        v = feat[(((size_t)b * 512 + c) * 50 + (py - 1)) * 50 + (px - 1)];
    padf[i] = (_Float16)v;
}

// =============== kernel 2: repack W1 -> [oc][tap*512+ic] f16 ===============
__global__ void repack_w1(const float* __restrict__ W1,
                          _Float16* __restrict__ Whn) {
    int i = blockIdx.x * blockDim.x + threadIdx.x;   // = oc*4608 + t*512 + ic
    if (i >= 512 * 4608) return;
    int ic = i & 511;
    int q  = i >> 9;        // oc*9 + t
    int t  = q % 9;
    int oc = q / 9;
    int ky = t / 3, kx = t % 3;
    Whn[i] = (_Float16)W1[(((size_t)oc * 512 + ic) * 3 + ky) * 3 + kx];
}

// =============== kernel 3: conv1 as implicit GEMM via WMMA ===============
union V16H { v16h v; unsigned u[8]; };

__device__ __forceinline__ v16h load_a_frag(const _Float16* sA, int rowbase, int lane) {
    // A frag, contiguous K order used here (same as B): lane row M=lane%16, K = g*16 + 0..15
    V16H r; const int g = lane >> 4;
    const _Float16* base = sA + (size_t)(rowbase + (lane & 15)) * 40;
#pragma unroll
    for (int i = 0; i < 8; ++i) {
        const int kk = (g << 4) + (i << 1);
        r.u[i] = *(const unsigned*)(base + kk);
    }
    return r.v;
}

__device__ __forceinline__ v16h load_b_frag(const _Float16* sBt, int colbase, int lane) {
    // B frag, contiguous K order used here (same as A): lane n holds col N=n%16, K = g*16 + 0..15
    V16H r; const int n = lane & 15, g = lane >> 4;
    const unsigned* pu = (const unsigned*)(sBt + (size_t)(colbase + n) * 40 + (g << 4));
#pragma unroll
    for (int i = 0; i < 8; ++i) r.u[i] = pu[i];
    return r.v;
}

__global__ __launch_bounds__(256)
void conv1_wmma(const _Float16* __restrict__ padf,
                const _Float16* __restrict__ Whn,
                const float* __restrict__ b1,
                _Float16* __restrict__ conv1h) {
    // double-buffered tiles: 2 x (128 rows/cols x 32 K, stride 40) = 40 KB LDS
    __shared__ __align__(16) _Float16 sA[2][128 * 40];
    __shared__ __align__(16) _Float16 sBt[2][128 * 40];

    const int tid  = threadIdx.x;
    const int lane = tid & 31;
    const int w    = tid >> 5;         // 8 waves
    const int wm   = w & 3;            // 4 M-groups of 32 rows
    const int wn   = w >> 2;           // 2 N-groups of 64 cols
    const int tile_m = blockIdx.x * 128;
    const int tile_n = blockIdx.y * 128;
    const int b      = blockIdx.z;

    v8f acc[2][4];
#pragma unroll
    for (int i = 0; i < 2; ++i)
#pragma unroll
        for (int j = 0; j < 4; ++j)
            acc[i][j] = (v8f){0.f, 0.f, 0.f, 0.f, 0.f, 0.f, 0.f, 0.f};

    // per-thread staging assignment (hoisted out of the K loop)
    const int ar = tid >> 1, ac = tid & 1;         // A: row 0..127, 16-half chunk
    const int m  = tile_m + ar;
    const int mc = (m < NPOS) ? m : (NPOS - 1);    // clamp; stores are guarded
    const int ay = mc / 50, ax = mc % 50;
    const _Float16* aBase = padf + (((size_t)b * 52 + ay) * 52 + ax) * 512 + (ac << 4);
    const int cc = tid >> 1, kk0 = (tid & 1) << 4; // B: col 0..127, 16-half chunk
    const _Float16* bBase = Whn + (size_t)(tile_n + cc) * 4608 + kk0;
    const int aLds = ar * 40 + (ac << 4);
    const int bLds = cc * 40 + kk0;

    auto issue_stage = [&](int kc, int bi) {
        const int t   = kc >> 4;                   // 3x3 tap (512 ic per tap)
        const int ic0 = (kc & 15) << 5;
        const int ky = t / 3, kx = t % 3;
        const _Float16* aSrc = aBase + ((ky * 52 + kx) << 9) + ic0;
        const _Float16* bSrc = bBase + (kc << 5);
        _Float16* aD = &sA[bi][aLds];
        _Float16* bD = &sBt[bi][bLds];
#if USE_ASYNC_LDS
        // CDNA5 async DMA: 32B per thread per matrix, ASYNCcnt-tracked,
        // immediate offset applies to both global and LDS addresses.
        __builtin_amdgcn_global_load_async_to_lds_b128(AS1P(aSrc), AS3P(aD), 0,  0);
        __builtin_amdgcn_global_load_async_to_lds_b128(AS1P(aSrc), AS3P(aD), 16, 0);
        __builtin_amdgcn_global_load_async_to_lds_b128(AS1P(bSrc), AS3P(bD), 0,  0);
        __builtin_amdgcn_global_load_async_to_lds_b128(AS1P(bSrc), AS3P(bD), 16, 0);
#else
        uint64_t av[4], bv[4];
#pragma unroll
        for (int q = 0; q < 4; ++q) av[q] = ((const uint64_t*)aSrc)[q];
#pragma unroll
        for (int q = 0; q < 4; ++q) bv[q] = ((const uint64_t*)bSrc)[q];
#pragma unroll
        for (int q = 0; q < 4; ++q) ((uint64_t*)aD)[q] = av[q];
#pragma unroll
        for (int q = 0; q < 4; ++q) ((uint64_t*)bD)[q] = bv[q];
#endif
    };

    issue_stage(0, 0);
    for (int kc = 0; kc < 144; ++kc) {             // K = 4608, step 32
        const int cur = kc & 1;
        if (kc + 1 < 144) {
            issue_stage(kc + 1, cur ^ 1);          // next tile in flight
#if USE_ASYNC_LDS
            __builtin_amdgcn_s_wait_asynccnt(4);   // current tile landed
#endif
        } else {
#if USE_ASYNC_LDS
            __builtin_amdgcn_s_wait_asynccnt(0);
#endif
        }
        __syncthreads();                           // all waves' copies visible

        v16h af[2], bf[4];
#pragma unroll
        for (int mt = 0; mt < 2; ++mt) af[mt] = load_a_frag(sA[cur], wm * 32 + mt * 16, lane);
#pragma unroll
        for (int nt = 0; nt < 4; ++nt) bf[nt] = load_b_frag(sBt[cur], wn * 64 + nt * 16, lane);
#pragma unroll
        for (int mt = 0; mt < 2; ++mt)
#pragma unroll
            for (int nt = 0; nt < 4; ++nt)
                acc[mt][nt] = __builtin_amdgcn_wmma_f32_16x16x32_f16(
                    false, af[mt], false, bf[nt], (short)0, acc[mt][nt], false, false);
        __syncthreads();                           // buffer free for reuse
    }

    // epilogue: + bias, ReLU, store f16 (C/D layout: VGPR v -> M=g*8+v, lane -> N)
    const int n = lane & 15, g = lane >> 4;
#pragma unroll
    for (int mt = 0; mt < 2; ++mt) {
#pragma unroll
        for (int nt = 0; nt < 4; ++nt) {
            const int col  = tile_n + wn * 64 + nt * 16 + n;
            const float bias = b1[col];
#pragma unroll
            for (int v = 0; v < 8; ++v) {
                const int row = tile_m + wm * 32 + mt * 16 + g * 8 + v;
                if (row < NPOS) {
                    const float x = acc[mt][nt][v] + bias;
                    conv1h[((size_t)b * NPOS + row) * 512 + col] = (_Float16)fmaxf(x, 0.0f);
                }
            }
        }
    }
}

// ======== kernel 4: 1x1 heads + softmax + anchor decode + size mask ========
__global__ __launch_bounds__(256)
void heads_decode(const _Float16* __restrict__ conv1h,
                  const float* __restrict__ Wreg, const float* __restrict__ breg,
                  const float* __restrict__ Wcls, const float* __restrict__ bcls,
                  float* __restrict__ boxes, float* __restrict__ scores) {
    __shared__ _Float16 sRow[4][512];
    __shared__ float    sOut[4][64];
    const int tid = threadIdx.x;
    const int gp0 = blockIdx.x * 4;                // 4 positions per block

    for (int i = tid; i < 4 * 512; i += 256) {
        const int lp = i >> 9, c = i & 511;
        sRow[lp][c] = conv1h[(size_t)(gp0 + lp) * 512 + c];
    }
    __syncthreads();

    const int lp = tid >> 6, oc = tid & 63;
    if (oc < 54) {                                 // 36 reg + 18 cls channels
        const float* wrow = (oc < 36) ? (Wreg + (size_t)oc * 512)
                                      : (Wcls + (size_t)(oc - 36) * 512);
        float s = (oc < 36) ? breg[oc] : bcls[oc - 36];
#pragma unroll 8
        for (int c = 0; c < 512; ++c) s += (float)sRow[lp][c] * wrow[c];
        sOut[lp][oc] = s;
    }
    __syncthreads();

    if (oc < 9) {
        const int a  = oc;
        const int gp = gp0 + lp;                   // global position, 0..19999
        const int b  = gp / NPOS, p = gp % NPOS;
        const int y  = p / 50, x = p % 50;
        const float d0 = sOut[lp][a * 4 + 0], d1 = sOut[lp][a * 4 + 1];
        const float d2 = sOut[lp][a * 4 + 2], d3 = sOut[lp][a * 4 + 3];
        const float s0 = sOut[lp][36 + a * 2], s1 = sOut[lp][36 + a * 2 + 1];
        const float fg = 1.0f / (1.0f + __expf(s0 - s1));   // softmax fg prob
        const float RR[3] = {0.5f, 1.0f, 2.0f};
        const float SS[3] = {8.0f, 16.0f, 32.0f};
        const float r  = RR[a / 3], sc = SS[a % 3];
        const float sq = sqrtf(r);
        const float ah = 16.0f * sc * sq;
        const float aw = 16.0f * sc / sq;
        const float acy = 16.0f * (float)(y + 1) - 25.0f;
        const float acx = 16.0f * (float)(x + 1) - 25.0f;
        const float cy = d0 * ah + acy, cx = d1 * aw + acx;
        const float h  = __expf(d2) * ah, wd = __expf(d3) * aw;
        const float y1 = fminf(fmaxf(cy - 0.5f * h,  0.f), 800.f);
        const float y2 = fminf(fmaxf(cy + 0.5f * h,  0.f), 800.f);
        const float x1 = fminf(fmaxf(cx - 0.5f * wd, 0.f), 800.f);
        const float x2 = fminf(fmaxf(cx + 0.5f * wd, 0.f), 800.f);
        const bool ok = ((y2 - y1) >= 16.0f) && ((x2 - x1) >= 16.0f);
        const size_t nidx = (size_t)b * NANCH + (size_t)p * 9 + a;
        boxes[nidx * 4 + 0] = y1; boxes[nidx * 4 + 1] = x1;
        boxes[nidx * 4 + 2] = y2; boxes[nidx * 4 + 3] = x2;
        scores[nidx] = ok ? fg : NEGV;
    }
}

// ============ kernel 5: greedy NMS, one workgroup per batch ============
__global__ __launch_bounds__(1024)
void nms_kernel(const float* __restrict__ boxes, float* __restrict__ live,
                float* __restrict__ out) {
    const int b = blockIdx.x, tid = threadIdx.x;
    const float* bx = boxes + (size_t)b * NANCH * 4;
    float* lv = live + (size_t)b * NANCH;
    __shared__ float rv[1024];
    __shared__ int   ridx[1024];
    __shared__ float sel[6];     // y1,x1,y2,x2,score,area
    __shared__ int   selv;
    float* rois = out + (size_t)b * POST * 5;
    float* rsc  = out + (size_t)BATCH * POST * 5 + (size_t)b * POST;

    for (int it = 0; it < POST; ++it) {
        float bv = -3.0e38f; int bi = 0;
        for (int i = tid; i < NANCH; i += 1024) {
            const float v = lv[i];
            if (v > bv || (v == bv && i < bi)) { bv = v; bi = i; }
        }
        rv[tid] = bv; ridx[tid] = bi;
        __syncthreads();
        for (int s = 512; s > 0; s >>= 1) {
            if (tid < s) {
                const float ov = rv[tid + s]; const int oi = ridx[tid + s];
                if (ov > rv[tid] || (ov == rv[tid] && oi < ridx[tid])) {
                    rv[tid] = ov; ridx[tid] = oi;
                }
            }
            __syncthreads();
        }
        if (tid == 0) {
            const int i = ridx[0]; const float sscore = rv[0];
            const float y1 = bx[i * 4 + 0], x1 = bx[i * 4 + 1];
            const float y2 = bx[i * 4 + 2], x2 = bx[i * 4 + 3];
            const int valid = sscore > (NEGV * 0.5f);
            sel[0] = y1; sel[1] = x1; sel[2] = y2; sel[3] = x2;
            sel[4] = sscore; sel[5] = (y2 - y1) * (x2 - x1);
            selv = valid;
            lv[i] = NEGV;
            if (valid) {
                rois[it * 5 + 0] = (float)b;
                rois[it * 5 + 1] = y1; rois[it * 5 + 2] = x1;
                rois[it * 5 + 3] = y2; rois[it * 5 + 4] = x2;
                rsc[it] = sscore;
            } else {
                rois[it * 5 + 0] = 0.f; rois[it * 5 + 1] = 0.f;
                rois[it * 5 + 2] = 0.f; rois[it * 5 + 3] = 0.f;
                rois[it * 5 + 4] = 0.f;
                rsc[it] = 0.f;
            }
        }
        __syncthreads();
        if (selv) {
            const float sy1 = sel[0], sx1 = sel[1], sy2 = sel[2], sx2 = sel[3];
            const float sar = sel[5];
            for (int i = tid; i < NANCH; i += 1024) {
                const float v = lv[i];
                if (v <= (NEGV * 0.5f)) continue;
                const float y1 = bx[i * 4 + 0], x1 = bx[i * 4 + 1];
                const float y2 = bx[i * 4 + 2], x2 = bx[i * 4 + 3];
                const float yy1 = fmaxf(sy1, y1), xx1 = fmaxf(sx1, x1);
                const float yy2 = fminf(sy2, y2), xx2 = fminf(sx2, x2);
                const float inter = fmaxf(yy2 - yy1, 0.f) * fmaxf(xx2 - xx1, 0.f);
                const float area  = (y2 - y1) * (x2 - x1);
                const float iou   = inter / (sar + area - inter + 1e-9f);
                if (iou > 0.7f) lv[i] = NEGV;
            }
        }
        __syncthreads();
    }
}

// ============================ launch ============================
extern "C" void kernel_launch(void* const* d_in, const int* in_sizes, int n_in,
                              void* d_out, int out_size, void* d_ws, size_t ws_size,
                              hipStream_t stream) {
    (void)in_sizes; (void)n_in; (void)out_size; (void)ws_size;
    const float* feat = (const float*)d_in[0];
    const float* W1   = (const float*)d_in[1];
    const float* b1   = (const float*)d_in[2];
    const float* Wreg = (const float*)d_in[3];
    const float* breg = (const float*)d_in[4];
    const float* Wcls = (const float*)d_in[5];
    const float* bcls = (const float*)d_in[6];
    float* out = (float*)d_out;
    char* ws = (char*)d_ws;
    _Float16* padf   = (_Float16*)(ws + OFF_PADF);
    _Float16* Whn    = (_Float16*)(ws + OFF_WHN);
    _Float16* conv1h = (_Float16*)(ws + OFF_CONV1);
    float* boxes     = (float*)(ws + OFF_BOXES);
    float* scores    = (float*)(ws + OFF_SCORES);

    {
        const size_t N = (size_t)BATCH * 52 * 52 * 512;
        pad_nhwc_f16<<<dim3((unsigned)((N + 255) / 256)), dim3(256), 0, stream>>>(feat, padf);
    }
    repack_w1<<<dim3((512 * 4608 + 255) / 256), dim3(256), 0, stream>>>(W1, Whn);
    conv1_wmma<<<dim3(20, 4, 8), dim3(256), 0, stream>>>(padf, Whn, b1, conv1h);
    heads_decode<<<dim3(20000 / 4), dim3(256), 0, stream>>>(conv1h, Wreg, breg,
                                                            Wcls, bcls, boxes, scores);
    nms_kernel<<<dim3(8), dim3(1024), 0, stream>>>(boxes, scores, out);
}
